// AttentionBasedNN_82532091559936
// MI455X (gfx1250) — matrix operand order repaired and checked by reading, not compiled, not run on hardware
//
#include <hip/hip_runtime.h>
#include <hip/hip_bf16.h>
#include <math.h>

typedef __attribute__((ext_vector_type(16))) _Float16 v16h;
typedef __attribute__((ext_vector_type(8)))  _Float16 v8h;
typedef __attribute__((ext_vector_type(8)))  float    v8f;

#define VEC 1280
#define HID 128
#define BATCH 4
#define SEQ 256
#define MROWS 1024   // BATCH*SEQ

static __device__ __forceinline__ v16h cat8(v8h x, v8h y)
{
    return __builtin_shufflevector(x, y, 0, 1, 2, 3, 4, 5, 6, 7,
                                         8, 9, 10, 11, 12, 13, 14, 15);
}

// LDS byte address of a __shared__ object (generic -> addrspace(3) cast).
typedef __attribute__((address_space(3))) _Float16 lds_h;
static __device__ __forceinline__ unsigned lds_addr_of(const _Float16* p)
{
    return (unsigned)(unsigned long long)(lds_h*)p;
}

// CDNA5 async global->LDS copy, 16 bytes per active lane (tracked by ASYNCcnt).
static __device__ __forceinline__ void async_copy16(unsigned lds_off,
                                                    const _Float16* gsrc)
{
    asm volatile("global_load_async_to_lds_b128 %0, %1, off"
                 :: "v"(lds_off), "v"(gsrc) : "memory");
}
static __device__ __forceinline__ void wait_async0()
{
    asm volatile("s_wait_asynccnt 0x0" ::: "memory");
}

// ---------------------------------------------------------------------------
// One-time converts: activations f32 -> f16 (+ transposed), weights f32[K,N]
// -> f16 WT[N,K] so WMMA B-fragments are contiguous.
// ---------------------------------------------------------------------------
__global__ void cvt_act_kernel(const float* __restrict__ src,
                               _Float16* __restrict__ dst,
                               _Float16* __restrict__ dstT)
{
    int id = blockIdx.x * 256 + threadIdx.x;          // 0 .. MROWS*VEC-1
    int r = id / VEC, c = id % VEC;
    _Float16 h = (_Float16)src[id];
    dst[id] = h;
    dstT[(size_t)c * MROWS + r] = h;
}

__global__ void cvt_wt_kernel(const float* __restrict__ W,
                              _Float16* __restrict__ WT, int K, int N)
{
    int id = blockIdx.x * 256 + threadIdx.x;          // 0 .. K*N-1
    int k = id / N, n = id % N;
    WT[(size_t)n * K + k] = (_Float16)W[id];
}

// ---------------------------------------------------------------------------
// Templated WMMA GEMM:  out = act(A[M,K](f16) @ W + bias) (+ resid)
// W given transposed as WT[N,K] (f16). grid (M/16, N/128), block 256 (8 waves).
// A tile (16x32 f16) staged to LDS with global_load_async_to_lds_b128.
// Optional outputs: f32 outF, f16 outH, f16 transposed outHT[N,M].
// ---------------------------------------------------------------------------
template <int M, int N, int K>
__global__ void gemm_wmma_kernel(const _Float16* __restrict__ A,
                                 const _Float16* __restrict__ WT,
                                 const float* __restrict__ bias,
                                 const float* __restrict__ resid,
                                 float* __restrict__ outF,
                                 _Float16* __restrict__ outH,
                                 _Float16* __restrict__ outHT,
                                 int relu)
{
    __shared__ _Float16 As[16 * 32];

    const int t    = threadIdx.x;
    const int wave = t >> 5;
    const int lane = t & 31;
    const int m0   = blockIdx.x * 16;
    const int n0   = blockIdx.y * 128 + wave * 16;

    const int row  = lane & 15;
    const int hi   = lane >> 4;
    const int kb   = hi * 16;
    const int col  = n0 + row;

    const unsigned lds_base = lds_addr_of(As);
    const int sr = t >> 2;              // staging row   (t < 64)
    const int sc = (t & 3) * 8;         // staging halves offset

    v8f acc = {};

    #pragma unroll 4
    for (int kk = 0; kk < K; kk += 32) {
        if (t < 64) {                   // waves 0-1 stage the shared A tile
            async_copy16(lds_base + (unsigned)((sr * 32 + sc) * 2),
                         A + (size_t)(m0 + sr) * K + kk + sc);
            wait_async0();
        }
        __syncthreads();

        v8h a0 = *(const v8h*)&As[row * 32 + kb];
        v8h a1 = *(const v8h*)&As[row * 32 + kb + 8];
        v16h a = cat8(a0, a1);
        v16h b = *(const v16h*)&WT[(size_t)col * K + kk + hi * 16];

        acc = __builtin_amdgcn_wmma_f32_16x16x32_f16(
            false, a, false, b, (short)0, acc, false, false);
        __syncthreads();
    }

    #pragma unroll
    for (int v = 0; v < 8; ++v) {
        int r = m0 + v + 8 * hi;
        float x = acc[v];
        if (bias)  x += bias[col];
        if (relu)  x = fmaxf(x, 0.0f);
        if (resid) x += resid[(size_t)r * N + col];
        if (outF)  outF[(size_t)r * N + col] = x;
        if (outH)  outH[(size_t)r * N + col] = (_Float16)x;
        if (outHT) outHT[(size_t)col * M + r] = (_Float16)x;
    }
}

// ---------------------------------------------------------------------------
// Additive attention (fused): scores = wv.tanh(q[i]+k[j]); softmax;
// Y = probs @ X + X (pre-LN, f32).  Values come from XT (f16, [VEC,MROWS]).
// grid (SEQ/16, BATCH), block 256.
// ---------------------------------------------------------------------------
__global__ void addattn_kernel(const float* __restrict__ Q,
                               const float* __restrict__ Kp,
                               const _Float16* __restrict__ XT,
                               const float* __restrict__ Xf,
                               const float* __restrict__ wv,
                               float* __restrict__ Y)
{
    __shared__ float    qs[16][HID];     // 8 KB
    __shared__ float    sc[16][SEQ];     // 16 KB
    __shared__ _Float16 ph[16][SEQ];     // 8 KB
    __shared__ float    wvs[HID];

    const int b  = blockIdx.y;
    const int m0 = blockIdx.x * 16;
    const int t  = threadIdx.x;
    const int wave = t >> 5, lane = t & 31;

    for (int i = t; i < 16 * HID; i += 256)
        qs[i / HID][i % HID] = Q[(size_t)(b * SEQ + m0 + i / HID) * HID + (i % HID)];
    if (t < HID) wvs[t] = wv[t];
    __syncthreads();

    {   // scores: thread -> row r = t%16, cols t/16 + 16*s
        const int r = t & 15, j0 = t >> 4;
        for (int s = 0; s < 16; ++s) {
            int j = j0 + 16 * s;
            const float* kr = &Kp[(size_t)(b * SEQ + j) * HID];
            float acc = 0.0f;
            #pragma unroll 4
            for (int h = 0; h < HID; ++h)
                acc += wvs[h] * tanhf(qs[r][h] + kr[h]);
            sc[r][j] = acc;
        }
    }
    __syncthreads();

    // per-row softmax on wave32 shuffles
    for (int rr = wave; rr < 16; rr += 8) {
        float mx = -INFINITY;
        for (int j = lane; j < SEQ; j += 32) mx = fmaxf(mx, sc[rr][j]);
        #pragma unroll
        for (int off = 16; off > 0; off >>= 1)
            mx = fmaxf(mx, __shfl_xor(mx, off, 32));
        float sum = 0.0f;
        for (int j = lane; j < SEQ; j += 32) {
            float e = __expf(sc[rr][j] - mx);
            sc[rr][j] = e;
            sum += e;
        }
        #pragma unroll
        for (int off = 16; off > 0; off >>= 1)
            sum += __shfl_xor(sum, off, 32);
        float inv = 1.0f / sum;
        for (int j = lane; j < SEQ; j += 32)
            ph[rr][j] = (_Float16)(sc[rr][j] * inv);
    }
    __syncthreads();

    // Y tile = probs(16xSEQ) @ V(SEQ x VEC) + X, WMMA over j
    const int row = lane & 15, hi = lane >> 4, kb = hi * 16;
    for (int nt = wave; nt < VEC / 16; nt += 8) {
        const int ncol = nt * 16 + row;
        const _Float16* vcol = &XT[(size_t)ncol * MROWS + b * SEQ];
        v8f acc = {};
        #pragma unroll
        for (int j = 0; j < SEQ; j += 32) {
            v8h a0 = *(const v8h*)&ph[row][j + kb];
            v8h a1 = *(const v8h*)&ph[row][j + kb + 8];
            v16h a  = cat8(a0, a1);
            v16h bf = *(const v16h*)&vcol[j + hi * 16];
            acc = __builtin_amdgcn_wmma_f32_16x16x32_f16(
                false, a, false, bf, (short)0, acc, false, false);
        }
        #pragma unroll
        for (int v = 0; v < 8; ++v) {
            int r = b * SEQ + m0 + v + 8 * hi;
            Y[(size_t)r * VEC + ncol] = acc[v] + Xf[(size_t)r * VEC + ncol];
        }
    }
}

// ---------------------------------------------------------------------------
// In-place row LayerNorm + f16 copy. grid = MROWS, block = 256.
// ---------------------------------------------------------------------------
__global__ void lnrow_kernel(float* __restrict__ Yio, _Float16* __restrict__ Yh)
{
    __shared__ float red[256];
    const int t = threadIdx.x;
    float* p = Yio + (size_t)blockIdx.x * VEC;
    _Float16* ph = Yh + (size_t)blockIdx.x * VEC;

    float s = 0.0f;
    for (int d = t; d < VEC; d += 256) s += p[d];
    red[t] = s; __syncthreads();
    for (int o = 128; o > 0; o >>= 1) { if (t < o) red[t] += red[t + o]; __syncthreads(); }
    const float mean = red[0] / VEC; __syncthreads();

    float v = 0.0f;
    for (int d = t; d < VEC; d += 256) { float x = p[d] - mean; v += x * x; }
    red[t] = v; __syncthreads();
    for (int o = 128; o > 0; o >>= 1) { if (t < o) red[t] += red[t + o]; __syncthreads(); }
    const float inv = rsqrtf(red[0] / VEC + 1e-5f); __syncthreads();

    for (int d = t; d < VEC; d += 256) {
        float x = (p[d] - mean) * inv;
        p[d]  = x;
        ph[d] = (_Float16)x;
    }
}

// ---------------------------------------------------------------------------
// Final: single-query attention at lys_pos + LN + MLP head. grid=BATCH.
// ---------------------------------------------------------------------------
__global__ void final_head_kernel(const float* __restrict__ Q4,
                                  const float* __restrict__ K4,
                                  const float* __restrict__ X3,
                                  const int* __restrict__ lys,
                                  const float* __restrict__ wv4,
                                  const float* __restrict__ hW1, const float* __restrict__ hb1,
                                  const float* __restrict__ hW2, const float* __restrict__ hb2,
                                  const float* __restrict__ hW3, const float* __restrict__ hb3,
                                  float* __restrict__ out)
{
    __shared__ float q[HID], sc[SEQ], red[256], y[VEC], h1[32], h2[12];
    const int b = blockIdx.x, t = threadIdx.x;
    const int L = lys[0];

    if (t < HID) q[t] = Q4[(size_t)(b * SEQ + L) * HID + t];
    __syncthreads();

    {
        const float* kr = &K4[(size_t)(b * SEQ + t) * HID];
        float a = 0.0f;
        #pragma unroll 4
        for (int h = 0; h < HID; ++h) a += wv4[h] * tanhf(q[h] + kr[h]);
        sc[t] = a;
    }
    __syncthreads();

    red[t] = sc[t]; __syncthreads();
    for (int o = 128; o > 0; o >>= 1) { if (t < o) red[t] = fmaxf(red[t], red[t + o]); __syncthreads(); }
    const float mx = red[0]; __syncthreads();
    float e = __expf(sc[t] - mx);
    sc[t] = e; red[t] = e; __syncthreads();
    for (int o = 128; o > 0; o >>= 1) { if (t < o) red[t] += red[t + o]; __syncthreads(); }
    const float inv = 1.0f / red[0]; __syncthreads();
    sc[t] *= inv; __syncthreads();

    for (int d = t; d < VEC; d += 256) {
        float a = 0.0f;
        for (int j = 0; j < SEQ; ++j) a += sc[j] * X3[(size_t)(b * SEQ + j) * VEC + d];
        y[d] = a + X3[(size_t)(b * SEQ + L) * VEC + d];
    }
    __syncthreads();

    float s = 0.0f;
    for (int d = t; d < VEC; d += 256) s += y[d];
    red[t] = s; __syncthreads();
    for (int o = 128; o > 0; o >>= 1) { if (t < o) red[t] += red[t + o]; __syncthreads(); }
    const float mean = red[0] / VEC; __syncthreads();
    float vv = 0.0f;
    for (int d = t; d < VEC; d += 256) { float x = y[d] - mean; vv += x * x; }
    red[t] = vv; __syncthreads();
    for (int o = 128; o > 0; o >>= 1) { if (t < o) red[t] += red[t + o]; __syncthreads(); }
    const float invs = rsqrtf(red[0] / VEC + 1e-5f); __syncthreads();
    for (int d = t; d < VEC; d += 256) y[d] = (y[d] - mean) * invs;
    __syncthreads();

    if (t < 32) {
        float a = hb1[t];
        for (int d = 0; d < VEC; ++d) a += y[d] * hW1[d * 32 + t];
        h1[t] = fmaxf(a, 0.0f);
    }
    __syncthreads();
    if (t < 12) {
        float a = hb2[t];
        #pragma unroll
        for (int d = 0; d < 32; ++d) a += h1[d] * hW2[d * 12 + t];
        h2[t] = fmaxf(a, 0.0f);
    }
    __syncthreads();
    if (t < 2) {
        float a = hb3[t];
        #pragma unroll
        for (int d = 0; d < 12; ++d) a += h2[d] * hW3[d * 2 + t];
        out[b * 2 + t] = a;
    }
}

// ---------------------------------------------------------------------------
extern "C" void kernel_launch(void* const* d_in, const int* in_sizes, int n_in,
                              void* d_out, int out_size, void* d_ws, size_t ws_size,
                              hipStream_t stream)
{
    (void)in_sizes; (void)n_in; (void)out_size; (void)ws_size;

    const float* X   = (const float*)d_in[0];
    const int*   lys = (const int*)d_in[1];
    const float *Wq[4], *Wk[4], *wv[4];
    for (int l = 0; l < 4; ++l) {
        Wq[l] = (const float*)d_in[2 + 3 * l];
        Wk[l] = (const float*)d_in[3 + 3 * l];
        wv[l] = (const float*)d_in[4 + 3 * l];
    }
    const float *rW1[3], *rb1[3], *rW2[3], *rb2[3];
    for (int l = 0; l < 3; ++l) {
        rW1[l] = (const float*)d_in[14 + 4 * l];
        rb1[l] = (const float*)d_in[15 + 4 * l];
        rW2[l] = (const float*)d_in[16 + 4 * l];
        rb2[l] = (const float*)d_in[17 + 4 * l];
    }
    const float* hW1 = (const float*)d_in[26];
    const float* hb1 = (const float*)d_in[27];
    const float* hW2 = (const float*)d_in[28];
    const float* hb2 = (const float*)d_in[29];
    const float* hW3 = (const float*)d_in[30];
    const float* hb3 = (const float*)d_in[31];

    // ---- workspace layout ----
    const size_t MV = (size_t)MROWS * VEC;       // 1,310,720
    const size_t MH = (size_t)MROWS * HID;       // 131,072
    const size_t WSZ = (size_t)VEC * HID;        // 163,840

    float* t0 = (float*)d_ws;          // [M,V] pre/post-LN
    float* t1 = t0 + MV;               // [M,V] layer out (f32)
    float* Qb = t1 + MV;               // [M,H]
    float* Kb = Qb + MH;               // [M,H]
    _Float16* hbase = (_Float16*)(Kb + MH);
    _Float16* Xh   = hbase;            // [M,V]
    _Float16* XhT  = Xh   + MV;        // [V,M]
    _Float16* t0h  = XhT  + MV;        // [M,V]  LN output f16
    _Float16* t1h  = t0h  + MV;        // [M,V]
    _Float16* t1hT = t1h  + MV;        // [V,M]
    _Float16* Hbh  = t1hT + MV;        // [M,H]  FC1 output f16
    _Float16* wt   = Hbh  + MH;        // 14 x [163840] transposed f16 weights

    _Float16* WqT[4], *WkT[4], *rW1T[3], *rW2T[3];
    for (int l = 0; l < 4; ++l) { WqT[l] = wt + (size_t)l * WSZ;
                                  WkT[l] = wt + (size_t)(4 + l) * WSZ; }
    for (int l = 0; l < 3; ++l) { rW1T[l] = wt + (size_t)(8 + l) * WSZ;
                                  rW2T[l] = wt + (size_t)(11 + l) * WSZ; }

    // ---- one-time conversions ----
    cvt_act_kernel<<<MV / 256, 256, 0, stream>>>(X, Xh, XhT);
    for (int l = 0; l < 4; ++l) {
        cvt_wt_kernel<<<WSZ / 256, 256, 0, stream>>>(Wq[l], WqT[l], VEC, HID);
        cvt_wt_kernel<<<WSZ / 256, 256, 0, stream>>>(Wk[l], WkT[l], VEC, HID);
    }
    for (int l = 0; l < 3; ++l) {
        cvt_wt_kernel<<<WSZ / 256, 256, 0, stream>>>(rW1[l], rW1T[l], VEC, HID);
        cvt_wt_kernel<<<WSZ / 256, 256, 0, stream>>>(rW2[l], rW2T[l], HID, VEC);
    }

    const dim3 gProj(MROWS / 16, 1);             // N=128
    const dim3 gWide(MROWS / 16, VEC / 128);     // N=1280
    const dim3 gAttn(SEQ / 16, BATCH);

    for (int l = 0; l < 3; ++l) {
        const _Float16* Ah  = (l == 0) ? Xh  : t1h;
        const _Float16* AhT = (l == 0) ? XhT : t1hT;
        const float*    Af  = (l == 0) ? X   : t1;

        gemm_wmma_kernel<MROWS, HID, VEC><<<gProj, 256, 0, stream>>>(
            Ah, WqT[l], nullptr, nullptr, Qb, nullptr, nullptr, 0);
        gemm_wmma_kernel<MROWS, HID, VEC><<<gProj, 256, 0, stream>>>(
            Ah, WkT[l], nullptr, nullptr, Kb, nullptr, nullptr, 0);
        addattn_kernel<<<gAttn, 256, 0, stream>>>(Qb, Kb, AhT, Af, wv[l], t0);
        lnrow_kernel<<<MROWS, 256, 0, stream>>>(t0, t0h);
        gemm_wmma_kernel<MROWS, HID, VEC><<<gProj, 256, 0, stream>>>(
            t0h, rW1T[l], rb1[l], nullptr, nullptr, Hbh, nullptr, 1);
        gemm_wmma_kernel<MROWS, VEC, HID><<<gWide, 256, 0, stream>>>(
            Hbh, rW2T[l], rb2[l], t0, t1, t1h, t1hT, 0);
    }

    gemm_wmma_kernel<MROWS, HID, VEC><<<gProj, 256, 0, stream>>>(
        t1h, WqT[3], nullptr, nullptr, Qb, nullptr, nullptr, 0);
    gemm_wmma_kernel<MROWS, HID, VEC><<<gProj, 256, 0, stream>>>(
        t1h, WkT[3], nullptr, nullptr, Kb, nullptr, nullptr, 0);
    final_head_kernel<<<BATCH, 256, 0, stream>>>(Qb, Kb, t1, lys, wv[3],
                                                 hW1, hb1, hW2, hb2, hW3, hb3,
                                                 (float*)d_out);
}
